// SymbolicLightModel_61332132986993
// MI455X (gfx1250) — matrix operand order repaired and checked by reading, not compiled, not run on hardware
//
#include <hip/hip_runtime.h>

// ---------------------------------------------------------------------------
// SymbolicLightModel on MI455X (gfx1250, wave32, WMMA).
// All GEMMs (tcam w/ow, ffn up/down, head) share one bf16 WMMA kernel:
//   out[m,n] = sum_k A[m,k] * W[n,k]      (A bf16 activations, W fp32 weights
//                                          converted to bf16 while staging LDS)
// v_wmma_f32_16x16x32_bf16, 128x64 block tile, 8 waves, 32x32 out per wave,
// double-buffered LDS (1 barrier / 64-K stage), global_prefetch of next stage.
// ---------------------------------------------------------------------------

#define S_LEN 2048
#define D_DIM 768
#define B_SZ  2
#define M_ROWS (B_SZ * S_LEN)   // 4096
#define I_DIM 2048
#define V_SZ  32000
#define L_NUM 12

typedef __bf16 bf16;
typedef __attribute__((ext_vector_type(16))) __bf16 v16bf;
typedef __attribute__((ext_vector_type(8)))  __bf16 v8bf;
typedef __attribute__((ext_vector_type(2)))  __bf16 v2bf;
typedef __attribute__((ext_vector_type(8)))  float  v8f;

#define LDS_STRIDE 72   // 64 bf16 + 8 pad (16B): keeps b128 alignment, kills bank conflicts

__device__ __forceinline__ unsigned pack2bf(float a, float b) {
  v2bf v;
  v.x = (__bf16)a;
  v.y = (__bf16)b;
  return __builtin_bit_cast(unsigned, v);   // -> v_cvt_pk_bf16_f32
}

__device__ __forceinline__ v16bf cat8(v8bf lo, v8bf hi) {
  return __builtin_shufflevector(lo, hi, 0,1,2,3,4,5,6,7,8,9,10,11,12,13,14,15);
}

// EPI: 0 = store f32, 1 = spike(v>=1) -> bf16, 2 = f32 + bias[n]
template<int EPI>
__global__ __launch_bounds__(256)
void wmma_gemm_kernel(const bf16* __restrict__ A, const float* __restrict__ W,
                      void* __restrict__ Out, const float* __restrict__ bias,
                      int M, int N, int K)
{
  __shared__ bf16 lA[2][128 * LDS_STRIDE];
  __shared__ bf16 lB[2][ 64 * LDS_STRIDE];

  const int tid   = threadIdx.x;
  const int lane  = tid & 31;
  const int wave  = tid >> 5;
  const int m_blk = blockIdx.x * 128;
  const int n_blk = blockIdx.y * 64;

  // staging: thread moves 32B of two A rows (srow, srow+64) and 16 floats of one W row
  const int srow = tid >> 2;         // 0..63
  const int scol = (tid & 3) * 16;   // 0,16,32,48

  // compute assignment: wave -> (mw 0..3, nw 0..1), each wave owns 32x32 of output
  const int mw    = (wave & 3) * 32;
  const int nw    = (wave >> 2) * 32;
  const int frow  = lane & 15;
  const int fhalf = lane >> 4;

  v8f acc[2][2] = {};

  const bf16*  aptr = A + (size_t)(m_blk + srow) * K + scol;  // + 64*K for second row
  const float* wptr = W + (size_t)(n_blk + srow) * K + scol;

  // ---- stage helper (manually inlined twice) --------------------------------
#define STAGE(BUF, KOFF)                                                        \
  {                                                                             \
    const uint4* s0 = (const uint4*)(aptr + (KOFF));                            \
    const uint4* s1 = (const uint4*)(aptr + (size_t)64 * K + (KOFF));           \
    uint4 a0 = s0[0], a1 = s0[1];                                               \
    uint4 b0 = s1[0], b1 = s1[1];                                               \
    uint4* d0 = (uint4*)&lA[BUF][srow * LDS_STRIDE + scol];                     \
    uint4* d1 = (uint4*)&lA[BUF][(srow + 64) * LDS_STRIDE + scol];              \
    d0[0] = a0; d0[1] = a1; d1[0] = b0; d1[1] = b1;                             \
    const float4* sw = (const float4*)(wptr + (KOFF));                          \
    float4 q0 = sw[0], q1 = sw[1], q2 = sw[2], q3 = sw[3];                      \
    uint4 p0, p1;                                                               \
    p0.x = pack2bf(q0.x, q0.y); p0.y = pack2bf(q0.z, q0.w);                     \
    p0.z = pack2bf(q1.x, q1.y); p0.w = pack2bf(q1.z, q1.w);                     \
    p1.x = pack2bf(q2.x, q2.y); p1.y = pack2bf(q2.z, q2.w);                     \
    p1.z = pack2bf(q3.x, q3.y); p1.w = pack2bf(q3.z, q3.w);                     \
    uint4* dw = (uint4*)&lB[BUF][srow * LDS_STRIDE + scol];                     \
    dw[0] = p0; dw[1] = p1;                                                     \
  }

  STAGE(0, 0)
  int buf = 0;

  for (int k0 = 0; k0 < K; k0 += 64) {
    __syncthreads();                      // staging of `buf` done; prev compute done
    if (k0 + 64 < K) {
      __builtin_prefetch(aptr + k0 + 128, 0, 0);            // global_prefetch_b8
      __builtin_prefetch(wptr + k0 + 128, 0, 0);
      STAGE(buf ^ 1, k0 + 64)
    }
    #pragma unroll
    for (int ks = 0; ks < 2; ++ks) {
      const int kk = ks * 32;
      v16bf bfrag[2], afrag[2];
      #pragma unroll
      for (int u = 0; u < 2; ++u) {
        // B frag (32x16): lane = N column, VGPR0-3 = K(half*8..+8), VGPR4-7 = K+16
        const bf16* bp = &lB[buf][(nw + u * 16 + frow) * LDS_STRIDE + kk + fhalf * 8];
        bfrag[u] = cat8(*(const v8bf*)bp, *(const v8bf*)(bp + 16));
      }
      #pragma unroll
      for (int t = 0; t < 2; ++t) {
        // A frag (16x32): lane = M row, VGPR0-3 = K(half*8..+8), VGPR4-7 = K+16
        const bf16* ap = &lA[buf][(mw + t * 16 + frow) * LDS_STRIDE + kk + fhalf * 8];
        afrag[t] = cat8(*(const v8bf*)ap, *(const v8bf*)(ap + 16));
      }
      #pragma unroll
      for (int t = 0; t < 2; ++t)
        #pragma unroll
        for (int u = 0; u < 2; ++u)
          acc[t][u] = __builtin_amdgcn_wmma_f32_16x16x32_bf16(
              false, afrag[t], false, bfrag[u], (short)0, acc[t][u], false, false);
    }
    buf ^= 1;
  }
#undef STAGE

  // ---- epilogue: C/D layout -> VGPR r holds M=r (lanes 0-15) / M=8+r (16-31)
  #pragma unroll
  for (int u = 0; u < 2; ++u) {
    const int n = n_blk + nw + u * 16 + frow;
    float bv = 0.f;
    if (EPI == 2) bv = bias[n];
    #pragma unroll
    for (int t = 0; t < 2; ++t) {
      const int mbase = m_blk + mw + t * 16 + fhalf * 8;
      #pragma unroll
      for (int r = 0; r < 8; ++r) {
        const size_t off = (size_t)(mbase + r) * N + n;
        const float v = acc[t][u][r];
        if (EPI == 0)      ((float*)Out)[off] = v;
        else if (EPI == 1) ((bf16*)Out)[off]  = (bf16)((v >= 1.0f) ? 1.0f : 0.0f);
        else               ((float*)Out)[off] = v + bv;
      }
    }
  }
}

// ---------------------------------------------------------------------------
// LayerNorm helpers: one block per (b,s) row, 256 threads, 3 elems/thread.
// ---------------------------------------------------------------------------
__device__ __forceinline__ void block_reduce2(float& a, float& b, float* red) {
  #pragma unroll
  for (int o = 16; o > 0; o >>= 1) { a += __shfl_down(a, o, 32); b += __shfl_down(b, o, 32); }
  const int lane = threadIdx.x & 31, w = threadIdx.x >> 5;
  if (lane == 0) { red[w] = a; red[8 + w] = b; }
  __syncthreads();
  if (threadIdx.x < 32) {
    a = (lane < 8) ? red[lane] : 0.f;
    b = (lane < 8) ? red[8 + lane] : 0.f;
    #pragma unroll
    for (int o = 4; o > 0; o >>= 1) { a += __shfl_down(a, o, 32); b += __shfl_down(b, o, 32); }
    if (lane == 0) { red[0] = a; red[8] = b; }
  }
  __syncthreads();
  a = red[0]; b = red[8];
}

__global__ __launch_bounds__(256)
void embed_ln_kernel(const int* __restrict__ tok, const float* __restrict__ emb,
                     const float* __restrict__ pos, const float* __restrict__ g,
                     const float* __restrict__ b, float* __restrict__ out)
{
  __shared__ float red[16];
  const int row = blockIdx.x;            // b*S + s
  const int s   = row & (S_LEN - 1);
  const int tid = threadIdx.x;
  const int t   = tok[row];
  const float* ep = emb + (size_t)t * D_DIM;
  const float* pp = pos + (size_t)s * D_DIM;
  float vals[3]; float sum = 0.f, sq = 0.f;
  #pragma unroll
  for (int i = 0; i < 3; ++i) {
    const int d = tid + i * 256;
    const float v = ep[d] + pp[d];
    vals[i] = v; sum += v; sq += v * v;
  }
  block_reduce2(sum, sq, red);
  const float mu   = sum * (1.0f / D_DIM);
  const float var  = sq * (1.0f / D_DIM) - mu * mu;
  const float rstd = rsqrtf(var + 1e-5f);
  #pragma unroll
  for (int i = 0; i < 3; ++i) {
    const int d = tid + i * 256;
    out[(size_t)row * D_DIM + d] = g[d] * (vals[i] - mu) * rstd + b[d];
  }
}

// residual + LN; writes f32 cont (in place), bf16 cont, optional bf16 spikes
template<bool EMIT_SPIKE>
__global__ __launch_bounds__(256)
void add_ln_kernel(float* __restrict__ cont, const float* __restrict__ delta,
                   const float* __restrict__ g, const float* __restrict__ b,
                   bf16* __restrict__ cont_bf, bf16* __restrict__ spikes)
{
  __shared__ float red[16];
  const int row = blockIdx.x;
  const int tid = threadIdx.x;
  const size_t base = (size_t)row * D_DIM;
  float vals[3]; float sum = 0.f, sq = 0.f;
  #pragma unroll
  for (int i = 0; i < 3; ++i) {
    const int d = tid + i * 256;
    const float v = cont[base + d] + delta[base + d];
    vals[i] = v; sum += v; sq += v * v;
  }
  block_reduce2(sum, sq, red);
  const float mu   = sum * (1.0f / D_DIM);
  const float var  = sq * (1.0f / D_DIM) - mu * mu;
  const float rstd = rsqrtf(var + 1e-5f);
  #pragma unroll
  for (int i = 0; i < 3; ++i) {
    const int d = tid + i * 256;
    const float xn = g[d] * (vals[i] - mu) * rstd + b[d];
    cont[base + d]    = xn;
    cont_bf[base + d] = (bf16)xn;
    if (EMIT_SPIKE) spikes[base + d] = (bf16)((xn >= 1.0f) ? 1.0f : 0.0f);
  }
}

// ---------------------------------------------------------------------------
// LIF recurrence over time: one thread per (b,d) column; loads pipelined 8-deep.
// ---------------------------------------------------------------------------
__global__ __launch_bounds__(256)
void lif_kernel(const float* __restrict__ x, bf16* __restrict__ spikes)
{
  const int idx = blockIdx.x * 256 + threadIdx.x;   // 0 .. B*D-1
  const int b = idx / D_DIM, d = idx - b * D_DIM;
  const float* xp = x + (size_t)b * S_LEN * D_DIM + d;
  bf16* sp = spikes + (size_t)b * S_LEN * D_DIM + d;
  float v = 0.f;
  for (int s0 = 0; s0 < S_LEN; s0 += 8) {
    float xs[8];
    #pragma unroll
    for (int i = 0; i < 8; ++i) xs[i] = xp[(size_t)(s0 + i) * D_DIM];
    #pragma unroll
    for (int i = 0; i < 8; ++i) {
      v = v * 0.95f + xs[i];
      const float s = (v >= 1.0f) ? 1.0f : 0.0f;
      sp[(size_t)(s0 + i) * D_DIM] = (bf16)s;
      v *= (1.0f - s);
    }
  }
}

// causal running mean: ctx[b,s,e] = (1/(s+1)) * sum_{s'<=s} t[b,s',e]  -> bf16
__global__ __launch_bounds__(256)
void cumsum_mean_kernel(const float* __restrict__ t, bf16* __restrict__ ctx)
{
  const int idx = blockIdx.x * 256 + threadIdx.x;
  const int b = idx / D_DIM, e = idx - b * D_DIM;
  const float* tp = t + (size_t)b * S_LEN * D_DIM + e;
  bf16* cp = ctx + (size_t)b * S_LEN * D_DIM + e;
  float acc = 0.f;
  for (int s0 = 0; s0 < S_LEN; s0 += 8) {
    float xs[8];
    #pragma unroll
    for (int i = 0; i < 8; ++i) xs[i] = tp[(size_t)(s0 + i) * D_DIM];
    #pragma unroll
    for (int i = 0; i < 8; ++i) {
      acc += xs[i];
      cp[(size_t)(s0 + i) * D_DIM] = (bf16)(acc / (float)(s0 + i + 1));
    }
  }
}

// ---------------------------------------------------------------------------
extern "C" void kernel_launch(void* const* d_in, const int* in_sizes, int n_in,
                              void* d_out, int out_size, void* d_ws, size_t ws_size,
                              hipStream_t stream)
{
  (void)in_sizes; (void)n_in; (void)out_size; (void)ws_size;
  const int*   tok       = (const int*)  d_in[0];
  const float* emb       = (const float*)d_in[1];
  const float* pos       = (const float*)d_in[2];
  const float* enc_g     = (const float*)d_in[3];
  const float* enc_b     = (const float*)d_in[4];
  const float* tcam_w    = (const float*)d_in[5];
  const float* tcam_ow   = (const float*)d_in[6];
  const float* tcam_g    = (const float*)d_in[7];
  const float* tcam_b    = (const float*)d_in[8];
  const float* ffn_up    = (const float*)d_in[9];
  const float* ffn_down  = (const float*)d_in[10];
  const float* ffn_g     = (const float*)d_in[11];
  const float* ffn_b     = (const float*)d_in[12];
  const float* head_w    = (const float*)d_in[13];
  const float* log_prior = (const float*)d_in[14];

  // workspace partition (~61 MB)
  char* w = (char*)d_ws;
  auto alloc = [&](size_t bytes) { char* p = w; w += (bytes + 255) & ~(size_t)255; return p; };
  float* contF  = (float*)alloc((size_t)M_ROWS * D_DIM * 4);  // x / cont (f32)
  float* gF     = (float*)alloc((size_t)M_ROWS * D_DIM * 4);  // t / out / h2 scratch
  bf16*  spikes = (bf16*) alloc((size_t)M_ROWS * D_DIM * 2);  // sp (layer input)
  bf16*  contB  = (bf16*) alloc((size_t)M_ROWS * D_DIM * 2);  // cont in bf16
  bf16*  ctxB   = (bf16*) alloc((size_t)M_ROWS * D_DIM * 2);  // running-mean ctx
  bf16*  hB     = (bf16*) alloc((size_t)M_ROWS * I_DIM * 2);  // spiked FFN hidden

  const dim3 blk(256);
  const int scanBlocks = (B_SZ * D_DIM) / 256;                // 6
  const int mGrid = M_ROWS / 128;                             // 32

  // encoder: embed + pos, LN
  embed_ln_kernel<<<M_ROWS, blk, 0, stream>>>(tok, emb, pos, enc_g, enc_b, contF);
  // LIF scan -> 0/1 spikes (bf16)
  lif_kernel<<<scanBlocks, blk, 0, stream>>>(contF, spikes);

  for (int l = 0; l < L_NUM; ++l) {
    const float* wl  = tcam_w   + (size_t)l * D_DIM * D_DIM;
    const float* owl = tcam_ow  + (size_t)l * D_DIM * D_DIM;
    const float* tg  = tcam_g   + (size_t)l * D_DIM;
    const float* tb  = tcam_b   + (size_t)l * D_DIM;
    const float* upl = ffn_up   + (size_t)l * I_DIM * D_DIM;
    const float* dwl = ffn_down + (size_t)l * D_DIM * I_DIM;
    const float* fg  = ffn_g    + (size_t)l * D_DIM;
    const float* fb  = ffn_b    + (size_t)l * D_DIM;

    // t = sp @ w^T   (mask is a no-op: sp >= 0, so sp*mask == sp)
    wmma_gemm_kernel<0><<<dim3(mGrid, D_DIM/64), blk, 0, stream>>>(
        spikes, wl, gF, nullptr, M_ROWS, D_DIM, D_DIM);
    // ctx = causal running mean of t
    cumsum_mean_kernel<<<scanBlocks, blk, 0, stream>>>(gF, ctxB);
    // out = ctx @ ow^T
    wmma_gemm_kernel<0><<<dim3(mGrid, D_DIM/64), blk, 0, stream>>>(
        ctxB, owl, gF, nullptr, M_ROWS, D_DIM, D_DIM);
    // cont2 = LN(cont + out); sp2 = spike(cont2)
    add_ln_kernel<true><<<M_ROWS, blk, 0, stream>>>(contF, gF, tg, tb, contB, spikes);
    // h = spike(cont2 @ up^T)
    wmma_gemm_kernel<1><<<dim3(mGrid, I_DIM/64), blk, 0, stream>>>(
        contB, upl, hB, nullptr, M_ROWS, I_DIM, D_DIM);
    // h2 = h @ down^T
    wmma_gemm_kernel<0><<<dim3(mGrid, D_DIM/64), blk, 0, stream>>>(
        hB, dwl, gF, nullptr, M_ROWS, D_DIM, I_DIM);
    // cont3 = LN(cont2 + h2)
    add_ln_kernel<false><<<M_ROWS, blk, 0, stream>>>(contF, gF, fg, fb, contB, nullptr);
  }

  // logits = cont @ head_w^T + log_prior  -> d_out (f32)
  wmma_gemm_kernel<2><<<dim3(mGrid, V_SZ/64), blk, 0, stream>>>(
      contB, head_w, d_out, log_prior, M_ROWS, V_SZ, D_DIM);
}
